// GATClassifier_1116691497588
// MI455X (gfx1250) — matrix operand order repaired and checked by reading, not compiled, not run on hardware
//
#include <hip/hip_runtime.h>
#include <hip/hip_bf16.h>

typedef __attribute__((ext_vector_type(16))) __bf16 v16bf;
typedef __attribute__((ext_vector_type(8)))  __bf16 v8bf;
typedef __attribute__((ext_vector_type(8)))  float  v8f;

#define NEG_SLOPE 0.2f
#define EPS_F 1e-16f

// ---------------------------------------------------------------- utilities

__device__ __forceinline__ void atomicMaxF(float* addr, float v) {
  unsigned int* ua = (unsigned int*)addr;
  unsigned int old = __float_as_uint(*addr);
  while (__uint_as_float(old) < v) {
    unsigned int assumed = old;
    old = atomicCAS(ua, assumed, __float_as_uint(v));
    if (old == assumed) break;
  }
}

__global__ void k_fill_f32(float* __restrict__ p, float v, long long n) {
  long long i = (long long)blockIdx.x * blockDim.x + threadIdx.x;
  if (i < n) p[i] = v;
}

__global__ void k_reduce_sum(const float* __restrict__ in, long long n,
                             float* __restrict__ out) {
  __shared__ float sm[256];
  float s = 0.f;
  for (long long i = (long long)blockIdx.x * blockDim.x + threadIdx.x; i < n;
       i += (long long)gridDim.x * blockDim.x)
    s += in[i];
  sm[threadIdx.x] = s;
  __syncthreads();
  for (int off = 128; off > 0; off >>= 1) {
    if ((int)threadIdx.x < off) sm[threadIdx.x] += sm[threadIdx.x + off];
    __syncthreads();
  }
  if (threadIdx.x == 0) atomicAdd(out, sm[0]);
}

__global__ void k_scalar_scale(float* __restrict__ p, float s) {
  if (blockIdx.x == 0 && threadIdx.x == 0) p[0] *= s;
}

// ce[h] = sum_c We[h*C+c] * att_e[h*C+c]   (ce pre-zeroed)
__global__ void k_edge_coeff(const float* __restrict__ We,
                             const float* __restrict__ att_e,
                             float* __restrict__ ce, int total, int C) {
  int t = blockIdx.x * blockDim.x + threadIdx.x;
  if (t < total) atomicAdd(&ce[t / C], We[t] * att_e[t]);
}

__global__ void k_f32_to_bf16(const float* __restrict__ in,
                              __bf16* __restrict__ out, long long n) {
  long long i = (long long)blockIdx.x * blockDim.x + threadIdx.x;
  if (i < n) out[i] = (__bf16)in[i];
}

// W[K,Ncol] row-major -> Wt[Ncol,K] row-major (bf16)
__global__ void k_transpose_bf16(const float* __restrict__ W,
                                 __bf16* __restrict__ Wt, int K, int Ncol) {
  long long i = (long long)blockIdx.x * blockDim.x + threadIdx.x;
  if (i >= (long long)K * Ncol) return;
  int k = (int)(i / Ncol), c = (int)(i - (long long)k * Ncol);
  Wt[(size_t)c * K + k] = (__bf16)W[i];
}

// ------------------------------------------------------------ WMMA GEMM
// C[M,Ncols] f32 = A[M,K] bf16 @ Bt[Ncols,K]^T bf16.  One wave per 16x16 tile.
// M % 16 == 0, K % 32 == 0, Ncols % 16 == 0.
__global__ __launch_bounds__(32) void k_wmma_gemm(
    const __bf16* __restrict__ A, const __bf16* __restrict__ Bt,
    float* __restrict__ Cm, int M, int K, int Ncols) {
  int ntilesN = Ncols >> 4;
  int tm = blockIdx.x / ntilesN;
  int tn = blockIdx.x - tm * ntilesN;
  int lane = threadIdx.x & 31;
  int half = lane >> 4;   // 0 or 1
  int l15  = lane & 15;

  const __bf16* arow = A  + (size_t)(tm * 16 + l15) * K;
  const __bf16* bcol = Bt + (size_t)(tn * 16 + l15) * K;

  v8f acc = {};
  for (int k0 = 0; k0 < K; k0 += 32) {
    // A 16x32 bf16 fragment: lanes 0-15 row M=l15, K = k0+{0..7,16..23};
    //                        lanes 16-31 row M=l15, K = k0+{8..15,24..31}.
    int ka = k0 + half * 8;
    v8bf a_lo = *(const v8bf*)(arow + ka);
    v8bf a_hi = *(const v8bf*)(arow + ka + 16);
    v16bf a = __builtin_shufflevector(a_lo, a_hi, 0, 1, 2, 3, 4, 5, 6, 7, 8, 9,
                                      10, 11, 12, 13, 14, 15);
    // B 32x16 bf16 fragment: lanes 0-15 col N=l15, K = k0+{0..7,16..23};
    //                        lanes 16-31 col N=l15, K = k0+{8..15,24..31}.
    int kb = k0 + half * 8;
    v8bf b_lo = *(const v8bf*)(bcol + kb);
    v8bf b_hi = *(const v8bf*)(bcol + kb + 16);
    v16bf b = __builtin_shufflevector(b_lo, b_hi, 0, 1, 2, 3, 4, 5, 6, 7, 8, 9,
                                      10, 11, 12, 13, 14, 15);
    acc = __builtin_amdgcn_wmma_f32_16x16x32_bf16(
        /*neg_a=*/false, a, /*neg_b=*/false, b,
        /*c_mod=*/(short)0, acc, /*reuse_a=*/false, /*reuse_b=*/false);
  }
  // C/D layout: lanes 0-15: N=l15, VGPR r -> M=r; lanes 16-31: M=r+8.
  int col  = tn * 16 + l15;
  int row0 = tm * 16 + half * 8;
#pragma unroll
  for (int r = 0; r < 8; ++r)
    Cm[(size_t)(row0 + r) * Ncols + col] = acc[r];
}

// ------------------------------------------------------- attention kernels

// asrc[n,h] = <feat[n,h,:], att_s[h,:]>, adst likewise.  idx over N*H.
__global__ void k_node_attn(const float* __restrict__ feat,
                            const float* __restrict__ att_s,
                            const float* __restrict__ att_d,
                            float* __restrict__ asrc, float* __restrict__ adst,
                            int N, int H, int C) {
  int idx = blockIdx.x * blockDim.x + threadIdx.x;
  if (idx >= N * H) return;
  int n = idx / H, h = idx - n * H;
  const float* row = feat + (size_t)n * H * C + (size_t)h * C;
  const float* as = att_s + h * C;
  const float* ad = att_d + h * C;
  float s = 0.f, d = 0.f;
  for (int c = 0; c < C; ++c) {
    s += row[c] * as[c];
    d += row[c] * ad[c];
  }
  asrc[idx] = s;
  adst[idx] = d;
}

__device__ __forceinline__ void edge_ends(const int* ei, int e, int E, int& s,
                                          int& d) {
  if (e < E) { s = ei[e]; d = ei[E + e]; }
  else       { s = e - E; d = e - E; }
}

// pass A: raw alpha (leaky relu) + segment max.  idx over EP*H.
__global__ void k_edge_alpha(const int* __restrict__ ei,
                             const float* __restrict__ eattr,
                             const float* __restrict__ ea_mean,
                             const float* __restrict__ ce,
                             const float* __restrict__ asrc,
                             const float* __restrict__ adst,
                             float* __restrict__ alpha,
                             float* __restrict__ nmax, int E, int EP, int H) {
  long long idx = (long long)blockIdx.x * blockDim.x + threadIdx.x;
  if (idx >= (long long)EP * H) return;
  int e = (int)(idx / H), h = (int)(idx - (long long)e * H);
  int s, d; edge_ends(ei, e, E, s, d);
  float ea = (e < E) ? eattr[e] : ea_mean[0];
  float a = asrc[s * H + h] + adst[d * H + h] + ea * ce[h];
  a = (a > 0.f) ? a : NEG_SLOPE * a;
  alpha[idx] = a;
  atomicMaxF(&nmax[d * H + h], a);
}

// pass B: exp(alpha - max) in place + segment sum.
__global__ void k_edge_exp(const int* __restrict__ ei,
                           const float* __restrict__ nmax,
                           float* __restrict__ alpha, float* __restrict__ den,
                           int E, int EP, int H) {
  long long idx = (long long)blockIdx.x * blockDim.x + threadIdx.x;
  if (idx >= (long long)EP * H) return;
  int e = (int)(idx / H), h = (int)(idx - (long long)e * H);
  int s, d; edge_ends(ei, e, E, s, d);
  float ex = expf(alpha[idx] - nmax[d * H + h]);
  alpha[idx] = ex;
  atomicAdd(&den[d * H + h], ex);
}

// pass B2: normalize in place.
__global__ void k_edge_norm(const int* __restrict__ ei,
                            const float* __restrict__ den,
                            float* __restrict__ alpha, int E, int EP, int H) {
  long long idx = (long long)blockIdx.x * blockDim.x + threadIdx.x;
  if (idx >= (long long)EP * H) return;
  int e = (int)(idx / H), h = (int)(idx - (long long)e * H);
  int s, d; edge_ends(ei, e, E, s, d);
  alpha[idx] = alpha[idx] / (den[d * H + h] + EPS_F);
}

// pass C: out[dst, j] += alpha[e,h] * feat[src, j].  idx over EP*HC, j=h*C+c.
__global__ void k_edge_agg(const int* __restrict__ ei,
                           const float* __restrict__ alpha,
                           const float* __restrict__ feat,
                           float* __restrict__ out, int E, int EP, int H,
                           int HC, int C) {
  long long idx = (long long)blockIdx.x * blockDim.x + threadIdx.x;
  if (idx >= (long long)EP * HC) return;
  int e = (int)(idx / HC), j = (int)(idx - (long long)e * HC);
  int h = j / C;
  int s, d; edge_ends(ei, e, E, s, d);
  float w = alpha[(long long)e * H + h];
  atomicAdd(&out[(size_t)d * HC + j], w * feat[(size_t)s * HC + j]);
}

// acc + bias -> elu -> bf16  (layer-1 output feeding GEMM2)
__global__ void k_bias_elu_bf16(const float* __restrict__ acc,
                                const float* __restrict__ bias,
                                __bf16* __restrict__ out, long long n, int HC) {
  long long i = (long long)blockIdx.x * blockDim.x + threadIdx.x;
  if (i >= n) return;
  int j = (int)(i % HC);
  float v = acc[i] + bias[j];
  v = (v > 0.f) ? v : (expf(v) - 1.f);
  out[i] = (__bf16)v;
}

// emb = elu(acc2 + b2); logits = emb@Wfc + bfc; log_softmax.
// d_out layout: [0, 2N) log-probs, [2N, 2N + N*C) emb.
__global__ void k_classify(const float* __restrict__ acc2,
                           const float* __restrict__ b2,
                           const float* __restrict__ Wfc,
                           const float* __restrict__ bfc,
                           float* __restrict__ out, int N, int C) {
  int n = blockIdx.x * blockDim.x + threadIdx.x;
  if (n >= N) return;
  float z0 = bfc[0], z1 = bfc[1];
  float* emb = out + (size_t)N * 2 + (size_t)n * C;
  const float* row = acc2 + (size_t)n * C;
  for (int c = 0; c < C; ++c) {
    float v = row[c] + b2[c];
    v = (v > 0.f) ? v : (expf(v) - 1.f);
    emb[c] = v;
    z0 += v * Wfc[c * 2 + 0];
    z1 += v * Wfc[c * 2 + 1];
  }
  float m = fmaxf(z0, z1);
  float lse = m + logf(expf(z0 - m) + expf(z1 - m));
  out[(size_t)n * 2 + 0] = z0 - lse;
  out[(size_t)n * 2 + 1] = z1 - lse;
}

// ---------------------------------------------------------------- launcher

static inline unsigned cdiv_ll(long long n, int b) {
  return (unsigned)((n + b - 1) / b);
}

extern "C" void kernel_launch(void* const* d_in, const int* in_sizes, int n_in,
                              void* d_out, int out_size, void* d_ws,
                              size_t ws_size, hipStream_t stream) {
  const float* x        = (const float*)d_in[0];
  const int*   ei       = (const int*)d_in[1];
  const float* eattr    = (const float*)d_in[2];
  const float* W1       = (const float*)d_in[3];
  const float* att_src1 = (const float*)d_in[4];
  const float* att_dst1 = (const float*)d_in[5];
  const float* We1      = (const float*)d_in[6];
  const float* att_e1   = (const float*)d_in[7];
  const float* b1       = (const float*)d_in[8];
  const float* W2       = (const float*)d_in[9];
  const float* att_src2 = (const float*)d_in[10];
  const float* att_dst2 = (const float*)d_in[11];
  const float* We2      = (const float*)d_in[12];
  const float* att_e2   = (const float*)d_in[13];
  const float* b2       = (const float*)d_in[14];
  const float* Wfc      = (const float*)d_in[15];
  const float* bfc      = (const float*)d_in[16];
  float* out = (float*)d_out;

  const int Fin = 128, H = 4, C = 64, HC = 256;
  const int N  = in_sizes[0] / Fin;   // 50000 (divisible by 16)
  const int E  = in_sizes[1] / 2;     // 800000
  const int EP = E + N;               // edges + self loops

  // ---- carve workspace
  size_t off = 0;
  auto carve = [&](size_t bytes) {
    size_t o = off;
    off += (bytes + 255) & ~(size_t)255;
    return (char*)d_ws + o;
  };
  __bf16* xb    = (__bf16*)carve((size_t)N * Fin * 2);
  __bf16* w1t   = (__bf16*)carve((size_t)HC * Fin * 2);
  __bf16* w2t   = (__bf16*)carve((size_t)C * HC * 2);
  float*  h1    = (float*)carve((size_t)N * HC * 4);
  float*  asrc1 = (float*)carve((size_t)N * H * 4);
  float*  adst1 = (float*)carve((size_t)N * H * 4);
  float*  alph1 = (float*)carve((size_t)EP * H * 4);
  float*  max1  = (float*)carve((size_t)N * H * 4);
  float*  den1  = (float*)carve((size_t)N * H * 4);
  float*  out1  = (float*)carve((size_t)N * HC * 4);
  __bf16* h1e   = (__bf16*)carve((size_t)N * HC * 2);
  float*  h2    = (float*)carve((size_t)N * C * 4);
  float*  asrc2 = (float*)carve((size_t)N * 4);
  float*  adst2 = (float*)carve((size_t)N * 4);
  float*  alph2 = (float*)carve((size_t)EP * 4);
  float*  max2  = (float*)carve((size_t)N * 4);
  float*  den2  = (float*)carve((size_t)N * 4);
  float*  out2  = (float*)carve((size_t)N * C * 4);
  float*  scal  = (float*)carve(8 * 4);  // [0]=ea_mean, [1..4]=ce1, [5]=ce2

  const int B = 256;
  const float NINF = -__builtin_inff();

  // ---- init
  k_fill_f32<<<1, 8, 0, stream>>>(scal, 0.f, 8);
  k_fill_f32<<<cdiv_ll((long long)N * H, B), B, 0, stream>>>(max1, NINF, (long long)N * H);
  k_fill_f32<<<cdiv_ll((long long)N * H, B), B, 0, stream>>>(den1, 0.f, (long long)N * H);
  k_fill_f32<<<cdiv_ll((long long)N * HC, B), B, 0, stream>>>(out1, 0.f, (long long)N * HC);
  k_fill_f32<<<cdiv_ll(N, B), B, 0, stream>>>(max2, NINF, N);
  k_fill_f32<<<cdiv_ll(N, B), B, 0, stream>>>(den2, 0.f, N);
  k_fill_f32<<<cdiv_ll((long long)N * C, B), B, 0, stream>>>(out2, 0.f, (long long)N * C);

  // ---- ea_mean and per-head edge coefficients
  k_reduce_sum<<<1024, B, 0, stream>>>(eattr, E, &scal[0]);
  k_scalar_scale<<<1, 32, 0, stream>>>(&scal[0], 1.f / (float)E);
  k_edge_coeff<<<1, HC, 0, stream>>>(We1, att_e1, &scal[1], HC, C);   // ce1[h]
  k_edge_coeff<<<1, C, 0, stream>>>(We2, att_e2, &scal[5], C, C);     // ce2[0]

  // ---- bf16 operands for WMMA
  k_f32_to_bf16<<<cdiv_ll((long long)N * Fin, B), B, 0, stream>>>(x, xb, (long long)N * Fin);
  k_transpose_bf16<<<cdiv_ll((long long)Fin * HC, B), B, 0, stream>>>(W1, w1t, Fin, HC);
  k_transpose_bf16<<<cdiv_ll((long long)HC * C, B), B, 0, stream>>>(W2, w2t, HC, C);

  // ---- layer 1: h1 = x @ W1  (WMMA bf16 -> f32)
  k_wmma_gemm<<<(N / 16) * (HC / 16), 32, 0, stream>>>(xb, w1t, h1, N, Fin, HC);

  k_node_attn<<<cdiv_ll((long long)N * H, B), B, 0, stream>>>(h1, att_src1, att_dst1, asrc1, adst1, N, H, C);

  k_edge_alpha<<<cdiv_ll((long long)EP * H, B), B, 0, stream>>>(
      ei, eattr, &scal[0], &scal[1], asrc1, adst1, alph1, max1, E, EP, H);
  k_edge_exp<<<cdiv_ll((long long)EP * H, B), B, 0, stream>>>(ei, max1, alph1, den1, E, EP, H);
  k_edge_norm<<<cdiv_ll((long long)EP * H, B), B, 0, stream>>>(ei, den1, alph1, E, EP, H);
  k_edge_agg<<<cdiv_ll((long long)EP * HC, B), B, 0, stream>>>(ei, alph1, h1, out1, E, EP, H, HC, C);

  k_bias_elu_bf16<<<cdiv_ll((long long)N * HC, B), B, 0, stream>>>(out1, b1, h1e, (long long)N * HC, HC);

  // ---- layer 2: h2 = elu(out1+b1) @ W2  (WMMA bf16 -> f32), H=1
  k_wmma_gemm<<<(N / 16) * (C / 16), 32, 0, stream>>>(h1e, w2t, h2, N, HC, C);

  k_node_attn<<<cdiv_ll(N, B), B, 0, stream>>>(h2, att_src2, att_dst2, asrc2, adst2, N, 1, C);

  k_edge_alpha<<<cdiv_ll(EP, B), B, 0, stream>>>(
      ei, eattr, &scal[0], &scal[5], asrc2, adst2, alph2, max2, E, EP, 1);
  k_edge_exp<<<cdiv_ll(EP, B), B, 0, stream>>>(ei, max2, alph2, den2, E, EP, 1);
  k_edge_norm<<<cdiv_ll(EP, B), B, 0, stream>>>(ei, den2, alph2, E, EP, 1);
  k_edge_agg<<<cdiv_ll((long long)EP * C, B), B, 0, stream>>>(ei, alph2, h2, out2, E, EP, 1, C, C);

  // ---- emb = elu(out2+b2); classifier + log_softmax
  k_classify<<<cdiv_ll(N, B), B, 0, stream>>>(out2, b2, Wfc, bfc, out, N, C);
}
